// QLinear_59906203844671
// MI455X (gfx1250) — matrix operand order repaired and checked by reading, not compiled, not run on hardware
//
#include <hip/hip_runtime.h>
#include <hip/hip_bf16.h>

typedef __attribute__((ext_vector_type(16))) __bf16 v16bf;
typedef __attribute__((ext_vector_type(8)))  float  v8f;
typedef __attribute__((ext_vector_type(4)))  unsigned int v4u;

union Frag16 { v16bf bf; v4u u[2]; };

static __device__ __forceinline__ unsigned short bfbits(__bf16 h) {
  union { __bf16 b; unsigned short u; } c; c.b = h; return c.u;
}
static __device__ __forceinline__ unsigned int pack2(__bf16 a, __bf16 b) {
  return (unsigned int)bfbits(a) | ((unsigned int)bfbits(b) << 16);
}

struct Split4 { unsigned int h0, h1, l0, l1; };

// Split one float4 (scaled by s) into packed-bf16 hi pairs and lo (residual) pairs.
static __device__ __forceinline__ Split4 split4(float4 f, float s) {
  float x0 = f.x * s, x1 = f.y * s, x2 = f.z * s, x3 = f.w * s;
  __bf16 b0 = (__bf16)x0, b1 = (__bf16)x1, b2 = (__bf16)x2, b3 = (__bf16)x3;
  Split4 r;
  r.h0 = pack2(b0, b1);
  r.h1 = pack2(b2, b3);
  r.l0 = pack2((__bf16)(x0 - (float)b0), (__bf16)(x1 - (float)b1));
  r.l1 = pack2((__bf16)(x2 - (float)b2), (__bf16)(x3 - (float)b3));
  return r;
}

#define TM 128
#define TN 128
#define TK 32
#define LDP 40   // padded LDS row stride (elements): 80B rows, conflict-free, 16B aligned
#define KSTEPS 64  // 2048 / TK

__global__ __launch_bounds__(256) void qlinear_wmma_bf16x2(
    const float* __restrict__ x,
    const float* __restrict__ rM,
    const float* __restrict__ iM,
    const float* __restrict__ jM,
    const float* __restrict__ kM,
    const float* __restrict__ bias,
    float* __restrict__ out)
{
  constexpr int K = 2048, N = 2048, Q = 512;

  __shared__ __attribute__((aligned(16))) __bf16 sXhi[TM][LDP];
  __shared__ __attribute__((aligned(16))) __bf16 sXlo[TM][LDP];
  __shared__ __attribute__((aligned(16))) __bf16 sBhi[TN][LDP];
  __shared__ __attribute__((aligned(16))) __bf16 sBlo[TN][LDP];

  const int tid  = threadIdx.x;
  const int lane = tid & 31;
  const int wave = tid >> 5;
  const int m0 = blockIdx.y * TM;
  const int n0 = blockIdx.x * TN;

  // Output quadrant (row of the Hamilton block matrix) — uniform per block.
  const int qo = n0 >> 9;
  const float *mr0, *mr1, *mr2, *mr3;
  float s0, s1, s2, s3;
  if (qo == 0)      { mr0=rM; mr1=iM; mr2=jM; mr3=kM; s0= 1.f; s1=-1.f; s2=-1.f; s3=-1.f; }
  else if (qo == 1) { mr0=iM; mr1=rM; mr2=kM; mr3=jM; s0= 1.f; s1= 1.f; s2=-1.f; s3= 1.f; }
  else if (qo == 2) { mr0=jM; mr1=kM; mr2=rM; mr3=iM; s0= 1.f; s1= 1.f; s2= 1.f; s3=-1.f; }
  else              { mr0=kM; mr1=jM; mr2=iM; mr3=rM; s0= 1.f; s1=-1.f; s2= 1.f; s3= 1.f; }

  v8f acc[8];
#pragma unroll
  for (int s = 0; s < 8; ++s)
#pragma unroll
    for (int v = 0; v < 8; ++v) acc[s][v] = 0.0f;

  const int lrow = tid >> 1;        // 0..127: tile row (X) / tile column n (B)
  const int lkh  = (tid & 1) * 16;  // 0 or 16: which half of the 32-wide K strip

  // Fragment-read coordinates (per ISA VGPR layouts)
  const int am  = (lane & 15) + 16 * wave;   // A: row within X tile
  const int akh = (lane < 16) ? 0 : 8;       // A: K base (v0-3 -> K 0-7 / 8-15)
  const int bnl = lane & 15;                 // B: column within 16-wide subtile
  const int bkh = (lane < 16) ? 0 : 8;       // B: K base (v0-3 -> K 0-7 / 8-15)

  const float* gxBase = x + (size_t)(m0 + lrow) * K + lkh;
  const size_t wRowOff = (size_t)((n0 & 511) + lrow) * Q + lkh;

  // ---- prologue: load tile 0 into registers ----
  float4 xr0, xr1, xr2, xr3, wr0, wr1, wr2, wr3;
  float sgCur = s0;
  {
    const float4* gx = (const float4*)(gxBase + 0);
    xr0 = gx[0]; xr1 = gx[1]; xr2 = gx[2]; xr3 = gx[3];
    const float4* gw = (const float4*)(mr0 + wRowOff + 0);
    wr0 = gw[0]; wr1 = gw[1]; wr2 = gw[2]; wr3 = gw[3];
  }

  for (int t = 0; t < KSTEPS; ++t) {
    // ---- convert + store tile t (already in registers) into LDS ----
    {
      Split4 p0 = split4(xr0, 1.0f);
      Split4 p1 = split4(xr1, 1.0f);
      Split4 p2 = split4(xr2, 1.0f);
      Split4 p3 = split4(xr3, 1.0f);
      v4u H0 = {p0.h0, p0.h1, p1.h0, p1.h1};
      v4u H1 = {p2.h0, p2.h1, p3.h0, p3.h1};
      v4u L0 = {p0.l0, p0.l1, p1.l0, p1.l1};
      v4u L1 = {p2.l0, p2.l1, p3.l0, p3.l1};
      *(v4u*)&sXhi[lrow][lkh]     = H0;
      *(v4u*)&sXhi[lrow][lkh + 8] = H1;
      *(v4u*)&sXlo[lrow][lkh]     = L0;
      *(v4u*)&sXlo[lrow][lkh + 8] = L1;
    }
    {
      Split4 p0 = split4(wr0, sgCur);
      Split4 p1 = split4(wr1, sgCur);
      Split4 p2 = split4(wr2, sgCur);
      Split4 p3 = split4(wr3, sgCur);
      v4u H0 = {p0.h0, p0.h1, p1.h0, p1.h1};
      v4u H1 = {p2.h0, p2.h1, p3.h0, p3.h1};
      v4u L0 = {p0.l0, p0.l1, p1.l0, p1.l1};
      v4u L1 = {p2.l0, p2.l1, p3.l0, p3.l1};
      *(v4u*)&sBhi[lrow][lkh]     = H0;
      *(v4u*)&sBhi[lrow][lkh + 8] = H1;
      *(v4u*)&sBlo[lrow][lkh]     = L0;
      *(v4u*)&sBlo[lrow][lkh + 8] = L1;
    }
    __syncthreads();   // tile t visible to all waves

    // ---- issue global loads for tile t+1; latency overlaps the WMMAs below ----
    if (t + 1 < KSTEPS) {
      const int k1 = (t + 1) * TK;
      const int qi = k1 >> 9;
      const float* Mp = (qi == 0) ? mr0 : (qi == 1) ? mr1 : (qi == 2) ? mr2 : mr3;
      sgCur           = (qi == 0) ? s0  : (qi == 1) ? s1  : (qi == 2) ? s2  : s3;
      const float4* gx = (const float4*)(gxBase + k1);
      xr0 = gx[0]; xr1 = gx[1]; xr2 = gx[2]; xr3 = gx[3];
      const float4* gw = (const float4*)(Mp + wRowOff + (k1 & 511));
      wr0 = gw[0]; wr1 = gw[1]; wr2 = gw[2]; wr3 = gw[3];
    }

    // ---- A fragments for this wave's 16-row strip (hi + lo) ----
    Frag16 ahi, alo;
    ahi.u[0] = *(const v4u*)&sXhi[am][akh];
    ahi.u[1] = *(const v4u*)&sXhi[am][akh + 16];
    alo.u[0] = *(const v4u*)&sXlo[am][akh];
    alo.u[1] = *(const v4u*)&sXlo[am][akh + 16];

#pragma unroll
    for (int sub = 0; sub < 8; ++sub) {
      Frag16 bhi, blo;
      bhi.u[0] = *(const v4u*)&sBhi[16 * sub + bnl][bkh];
      bhi.u[1] = *(const v4u*)&sBhi[16 * sub + bnl][bkh + 16];
      blo.u[0] = *(const v4u*)&sBlo[16 * sub + bnl][bkh];
      blo.u[1] = *(const v4u*)&sBlo[16 * sub + bnl][bkh + 16];
      acc[sub] = __builtin_amdgcn_wmma_f32_16x16x32_bf16(
          false, ahi.bf, false, bhi.bf, (short)0, acc[sub], false, false);
      acc[sub] = __builtin_amdgcn_wmma_f32_16x16x32_bf16(
          false, ahi.bf, false, blo.bf, (short)0, acc[sub], false, false);
      acc[sub] = __builtin_amdgcn_wmma_f32_16x16x32_bf16(
          false, alo.bf, false, bhi.bf, (short)0, acc[sub], false, false);
    }
    __syncthreads();   // all waves done reading tile t
  }

  // ---- epilogue: bias add + store (C layout: lanes 0-15 M=v, lanes 16-31 M=v+8)
  const int cn = lane & 15;
  const int mb = m0 + 16 * wave + ((lane < 16) ? 0 : 8);
#pragma unroll
  for (int sub = 0; sub < 8; ++sub) {
    const int n = n0 + 16 * sub + cn;
    const float bv = bias[n];
#pragma unroll
    for (int v = 0; v < 8; ++v)
      out[(size_t)(mb + v) * N + n] = acc[sub][v] + bv;
  }
}

extern "C" void kernel_launch(void* const* d_in, const int* in_sizes, int n_in,
                              void* d_out, int out_size, void* d_ws, size_t ws_size,
                              hipStream_t stream) {
  const float* x  = (const float*)d_in[0];
  const float* r  = (const float*)d_in[1];
  const float* i_ = (const float*)d_in[2];
  const float* j_ = (const float*)d_in[3];
  const float* k_ = (const float*)d_in[4];
  const float* b  = (const float*)d_in[5];
  float* out = (float*)d_out;

  // M=16384, N=2048 -> grid (N/128, M/128)
  dim3 grid(2048 / TN, 16384 / TM);
  dim3 block(256);
  hipLaunchKernelGGL(qlinear_wmma_bf16x2, grid, block, 0, stream,
                     x, r, i_, j_, k_, b, out);
}
